// PointNetFeaturePropagationDynamic_38268158607593
// MI455X (gfx1250) — matrix operand order repaired and checked by reading, not compiled, not run on hardware
//
#include <hip/hip_runtime.h>
#include <hip/hip_bf16.h>

// ---------------------------------------------------------------------------
// Problem constants (match reference)
// ---------------------------------------------------------------------------
#define NQ   16384   // N query points
#define NS   4096    // S source points
#define D1C  256
#define D2C  256
#define CIN  512     // D1+D2
#define COUT 256

typedef __attribute__((ext_vector_type(16))) _Float16 v16h;
typedef __attribute__((ext_vector_type(8)))  _Float16 v8h;
typedef __attribute__((ext_vector_type(8)))  float    v8f;

// ---------------------------------------------------------------------------
// Workspace layout (bytes, all 256B aligned)
// ---------------------------------------------------------------------------
#define OFF_IDX   ((size_t)0)          // int   [NQ*3]
#define OFF_WGT   ((size_t)196608)     // float [NQ*3]
#define OFF_P2T   ((size_t)393216)     // float [NS][256]   points2 transposed
#define OFF_W0H   ((size_t)4587520)    // f16   [256][512]
#define OFF_W1H   ((size_t)4849664)    // f16   [256][256]
#define OFF_STATS ((size_t)4980736)    // float [4*256] mean0,rstd0,mean1,rstd1
#define OFF_FH    ((size_t)4984832)    // f16   [NQ][512]   layer-1 B operand
#define OFF_X2H   ((size_t)21762048)   // f16   [NQ][256]   layer-2 B operand
#define OFF_Y     ((size_t)30150656)   // float [256][NQ]   pre-BN activations

// ---------------------------------------------------------------------------
// f32 -> f16 convert (weights)
// ---------------------------------------------------------------------------
__global__ __launch_bounds__(256) void cvt_f16_kernel(const float* __restrict__ src,
                                                      _Float16* __restrict__ dst, int n) {
  int i = blockIdx.x * 256 + threadIdx.x;
  if (i < n) dst[i] = (_Float16)src[i];
}

// ---------------------------------------------------------------------------
// Transpose f32 [256][cols] -> [cols][256], 64x64 LDS tiles (conflict-free)
// grid: (cols/64, 256/64), block 256
// ---------------------------------------------------------------------------
__global__ __launch_bounds__(256) void transpose_f32_kernel(const float* __restrict__ src,
                                                            float* __restrict__ dst, int cols) {
  __shared__ float s[64 * 65];
  const int n0 = blockIdx.x * 64;
  const int m0 = blockIdx.y * 64;
  for (int f = threadIdx.x; f < 4096; f += 256) {
    int mm = f >> 6, nn = f & 63;
    s[mm * 65 + nn] = src[(size_t)(m0 + mm) * cols + n0 + nn];
  }
  __syncthreads();
  for (int f = threadIdx.x; f < 4096; f += 256) {
    int nn = f >> 6, mm = f & 63;
    dst[(size_t)(n0 + nn) * 256 + m0 + mm] = s[mm * 65 + nn];
  }
}

// ---------------------------------------------------------------------------
// Transpose f32 [256][cols] -> f16 rows dst[(n)*dstStride + dstOff + m]
// (used for points1 -> feature rows)
// ---------------------------------------------------------------------------
__global__ __launch_bounds__(256) void transpose_f16_kernel(const float* __restrict__ src,
                                                            _Float16* __restrict__ dst,
                                                            int cols, int dstStride, int dstOff) {
  __shared__ float s[64 * 65];
  const int n0 = blockIdx.x * 64;
  const int m0 = blockIdx.y * 64;
  for (int f = threadIdx.x; f < 4096; f += 256) {
    int mm = f >> 6, nn = f & 63;
    s[mm * 65 + nn] = src[(size_t)(m0 + mm) * cols + n0 + nn];
  }
  __syncthreads();
  for (int f = threadIdx.x; f < 4096; f += 256) {
    int nn = f >> 6, mm = f & 63;
    dst[(size_t)(n0 + nn) * dstStride + dstOff + m0 + mm] = (_Float16)s[mm * 65 + nn];
  }
}

// ---------------------------------------------------------------------------
// Masked 3-NN: one thread per query, xyz2/id2 staged in LDS (broadcast reads).
// Branchless top-3 insertion network. grid: NQ/256, block 256.
// ---------------------------------------------------------------------------
__global__ __launch_bounds__(256) void knn_kernel(const float* __restrict__ xyz1,
                                                  const float* __restrict__ xyz2,
                                                  const int* __restrict__ pid1,
                                                  const int* __restrict__ pid2,
                                                  int* __restrict__ idxOut,
                                                  float* __restrict__ wOut) {
  __shared__ float sx[NS];
  __shared__ float sy[NS];
  __shared__ float sz[NS];
  __shared__ unsigned char sid[NS];
  for (int i = threadIdx.x; i < NS; i += 256) {
    sx[i] = xyz2[i];
    sy[i] = xyz2[NS + i];
    sz[i] = xyz2[2 * NS + i];
    sid[i] = (unsigned char)pid2[i];
  }
  __syncthreads();

  const int q = blockIdx.x * 256 + threadIdx.x;
  const float px = xyz1[q], py = xyz1[NQ + q], pz = xyz1[2 * NQ + q];
  const int pid = pid1[q];

  const float SENT = 3.0e38f;
  float d0 = SENT, d1 = SENT, d2 = SENT;
  int i0 = 0, i1 = 0, i2 = 0;

  for (int j = 0; j < NS; ++j) {
    float dx = px - sx[j];
    float dy = py - sy[j];
    float dz = pz - sz[j];
    float d = fmaf(dz, dz, fmaf(dy, dy, dx * dx));
    d = ((int)sid[j] == pid) ? d : SENT;
    bool c2 = d < d2, c1 = d < d1, c0 = d < d0;
    float t2d = c1 ? d1 : d;  int t2i = c1 ? i1 : j;
    d2 = c2 ? t2d : d2;       i2 = c2 ? t2i : i2;
    float t1d = c0 ? d0 : d;  int t1i = c0 ? i0 : j;
    d1 = c1 ? t1d : d1;       i1 = c1 ? t1i : i1;
    d0 = c0 ? d  : d0;        i0 = c0 ? j   : i0;
  }

  // fill_value semantics: missing neighbor -> reuse i0 with huge distance
  const float TH = 1.0e37f, BIGD = 1.0e8f;
  if (d0 > TH) { d0 = BIGD; }
  if (d1 > TH) { d1 = BIGD; i1 = i0; }
  if (d2 > TH) { d2 = BIGD; i2 = i0; }
  float w0 = 1.0f / (d0 + 1e-8f);
  float w1 = 1.0f / (d1 + 1e-8f);
  float w2 = 1.0f / (d2 + 1e-8f);
  float inv = 1.0f / (w0 + w1 + w2);
  idxOut[q * 3 + 0] = i0;  wOut[q * 3 + 0] = w0 * inv;
  idxOut[q * 3 + 1] = i1;  wOut[q * 3 + 1] = w1 * inv;
  idxOut[q * 3 + 2] = i2;  wOut[q * 3 + 2] = w2 * inv;
}

// ---------------------------------------------------------------------------
// Interpolate: contiguous row-gathers from P2t [NS][256]; writes f16 into
// the second half of each feature row Fh[q][256..512). grid NQ/32, block 256.
// Thread t owns channel c = t; loop over 32 queries (uniform idx/w loads).
// ---------------------------------------------------------------------------
__global__ __launch_bounds__(256) void interp_kernel(const float* __restrict__ P2t,
                                                     const int* __restrict__ idxb,
                                                     const float* __restrict__ wgt,
                                                     _Float16* __restrict__ Fh) {
  const int c = threadIdx.x;
  const int n0 = blockIdx.x * 32;
  for (int k = 0; k < 32; ++k) {
    int q = n0 + k;
    int i0 = idxb[q * 3 + 0], i1 = idxb[q * 3 + 1], i2 = idxb[q * 3 + 2];
    float w0 = wgt[q * 3 + 0], w1 = wgt[q * 3 + 1], w2 = wgt[q * 3 + 2];
    float v = w0 * P2t[(size_t)i0 * 256 + c]
            + w1 * P2t[(size_t)i1 * 256 + c]
            + w2 * P2t[(size_t)i2 * 256 + c];
    Fh[(size_t)q * CIN + D1C + c] = (_Float16)v;
  }
}

// ---------------------------------------------------------------------------
// WMMA GEMM: C[M][Nn] = A[M][K] * B^T (B stored as [Nn][K] f16 rows).
// One wave computes a 64x16 C strip (4 M-tiles) so each B fragment is reused
// by 4 independent v_wmma_f32_16x16x32_f16 ops (4x less feature traffic).
// A frag (16x32 f16): lane<16 holds row m K0..15; lane>=16 K16..31.
// B frag (32x16 f16): lane<16 holds col n K0..15; lane>=16 K16..31.
// grid: (Nn/16, M/64), block 32.
// ---------------------------------------------------------------------------
__global__ __launch_bounds__(32) void wmma_gemm_kernel(const _Float16* __restrict__ A,
                                                       const _Float16* __restrict__ B,
                                                       float* __restrict__ C,
                                                       int K, int Nn) {
  const int n0 = blockIdx.x * 16;
  const int m0 = blockIdx.y * 64;
  const int l = threadIdx.x;
  const int half = l >> 4;
  const int lane16 = l & 15;

  const _Float16* Brow = B + (size_t)(n0 + lane16) * K + half * 16;
  const _Float16* A0 = A + (size_t)(m0 + lane16) * K;
  const size_t astep = (size_t)16 * K;
  const int aoff0 = half ? 16 : 0;
  const int aoff1 = half ? 24 : 8;

  v8f acc0 = {}, acc1 = {}, acc2 = {}, acc3 = {};
  for (int kk = 0; kk < K; kk += 32) {
    v16h b = *(const v16h*)(Brow + kk);
    union { v16h v; v8h h[2]; } a0, a1, a2, a3;
    a0.h[0] = *(const v8h*)(A0 + kk + aoff0);
    a0.h[1] = *(const v8h*)(A0 + kk + aoff1);
    a1.h[0] = *(const v8h*)(A0 + astep + kk + aoff0);
    a1.h[1] = *(const v8h*)(A0 + astep + kk + aoff1);
    a2.h[0] = *(const v8h*)(A0 + 2 * astep + kk + aoff0);
    a2.h[1] = *(const v8h*)(A0 + 2 * astep + kk + aoff1);
    a3.h[0] = *(const v8h*)(A0 + 3 * astep + kk + aoff0);
    a3.h[1] = *(const v8h*)(A0 + 3 * astep + kk + aoff1);
    acc0 = __builtin_amdgcn_wmma_f32_16x16x32_f16(false, a0.v, false, b, (short)0, acc0, false, false);
    acc1 = __builtin_amdgcn_wmma_f32_16x16x32_f16(false, a1.v, false, b, (short)0, acc1, false, false);
    acc2 = __builtin_amdgcn_wmma_f32_16x16x32_f16(false, a2.v, false, b, (short)0, acc2, false, false);
    acc3 = __builtin_amdgcn_wmma_f32_16x16x32_f16(false, a3.v, false, b, (short)0, acc3, false, false);
  }
#pragma unroll
  for (int r = 0; r < 8; ++r) {
    int mrow = r + half * 8;
    int col = n0 + lane16;
    C[(size_t)(m0 + mrow) * Nn + col]      = acc0[r];
    C[(size_t)(m0 + 16 + mrow) * Nn + col] = acc1[r];
    C[(size_t)(m0 + 32 + mrow) * Nn + col] = acc2[r];
    C[(size_t)(m0 + 48 + mrow) * Nn + col] = acc3[r];
  }
}

// ---------------------------------------------------------------------------
// Per-channel batch-norm statistics over Nn samples. grid: 256 (one block per
// channel), block 256.
// ---------------------------------------------------------------------------
__global__ __launch_bounds__(256) void bn_stats_kernel(const float* __restrict__ Y,
                                                       float* __restrict__ mean,
                                                       float* __restrict__ rstd,
                                                       int Nn) {
  __shared__ float ssum[256];
  __shared__ float ssq[256];
  const int m = blockIdx.x;
  const int t = threadIdx.x;
  const float* row = Y + (size_t)m * Nn;
  float s = 0.0f, s2 = 0.0f;
  for (int n = t; n < Nn; n += 256) {
    float v = row[n];
    s += v;
    s2 = fmaf(v, v, s2);
  }
  ssum[t] = s; ssq[t] = s2;
  __syncthreads();
  for (int off = 128; off > 0; off >>= 1) {
    if (t < off) { ssum[t] += ssum[t + off]; ssq[t] += ssq[t + off]; }
    __syncthreads();
  }
  if (t == 0) {
    float mu = ssum[0] / (float)Nn;
    float var = ssq[0] / (float)Nn - mu * mu;
    var = var < 0.0f ? 0.0f : var;
    mean[m] = mu;
    rstd[m] = rsqrtf(var + 1e-5f);
  }
}

// ---------------------------------------------------------------------------
// BN + affine + ReLU, then transpose [256][Nn] f32 -> [Nn][256] f16 rows
// (B operand for the next GEMM). 64x64 LDS tiles. grid: (Nn/64, 4), block 256.
// ---------------------------------------------------------------------------
__global__ __launch_bounds__(256) void bn_act_transpose_kernel(const float* __restrict__ Y,
                                                               const float* __restrict__ mean,
                                                               const float* __restrict__ rstd,
                                                               const float* __restrict__ g,
                                                               const float* __restrict__ bt,
                                                               _Float16* __restrict__ Xh,
                                                               int Nn) {
  __shared__ float s[64 * 65];
  const int n0 = blockIdx.x * 64;
  const int m0 = blockIdx.y * 64;
  for (int f = threadIdx.x; f < 4096; f += 256) {
    int mm = f >> 6, nn = f & 63;
    int m = m0 + mm;
    float v = Y[(size_t)m * Nn + n0 + nn];
    v = (v - mean[m]) * rstd[m] * g[m] + bt[m];
    s[mm * 65 + nn] = v > 0.0f ? v : 0.0f;
  }
  __syncthreads();
  for (int f = threadIdx.x; f < 4096; f += 256) {
    int nn = f >> 6, mm = f & 63;
    Xh[(size_t)(n0 + nn) * 256 + m0 + mm] = (_Float16)s[mm * 65 + nn];
  }
}

// ---------------------------------------------------------------------------
// Final BN + affine + ReLU, f32 output in [1,256,N] layout. grid: 256*NQ/256.
// ---------------------------------------------------------------------------
__global__ __launch_bounds__(256) void bn_act_out_kernel(const float* __restrict__ Y,
                                                         const float* __restrict__ mean,
                                                         const float* __restrict__ rstd,
                                                         const float* __restrict__ g,
                                                         const float* __restrict__ bt,
                                                         float* __restrict__ outp) {
  size_t idx = (size_t)blockIdx.x * 256 + threadIdx.x;
  int m = (int)(idx >> 14);  // / NQ
  float v = Y[idx];
  v = (v - mean[m]) * rstd[m] * g[m] + bt[m];
  outp[idx] = v > 0.0f ? v : 0.0f;
}

// ---------------------------------------------------------------------------
// Host-side orchestration
// ---------------------------------------------------------------------------
extern "C" void kernel_launch(void* const* d_in, const int* in_sizes, int n_in,
                              void* d_out, int out_size, void* d_ws, size_t ws_size,
                              hipStream_t stream) {
  const float* xyz1    = (const float*)d_in[0];
  const float* xyz2    = (const float*)d_in[1];
  const int*   pid1    = (const int*)d_in[2];
  const int*   pid2    = (const int*)d_in[3];
  const float* points1 = (const float*)d_in[4];
  const float* points2 = (const float*)d_in[5];
  const float* w0      = (const float*)d_in[6];
  const float* g0      = (const float*)d_in[8];
  const float* bt0     = (const float*)d_in[9];
  const float* w1      = (const float*)d_in[10];
  const float* g1      = (const float*)d_in[12];
  const float* bt1     = (const float*)d_in[13];
  (void)in_sizes; (void)n_in; (void)out_size; (void)ws_size;

  char* ws = (char*)d_ws;
  int*      idxb  = (int*)(ws + OFF_IDX);
  float*    wgt   = (float*)(ws + OFF_WGT);
  float*    P2t   = (float*)(ws + OFF_P2T);
  _Float16* W0h   = (_Float16*)(ws + OFF_W0H);
  _Float16* W1h   = (_Float16*)(ws + OFF_W1H);
  float*    stats = (float*)(ws + OFF_STATS);
  _Float16* Fh    = (_Float16*)(ws + OFF_FH);
  _Float16* X2h   = (_Float16*)(ws + OFF_X2H);
  float*    Y     = (float*)(ws + OFF_Y);
  float* mean0 = stats, *rstd0 = stats + 256, *mean1 = stats + 512, *rstd1 = stats + 768;
  float* outF = (float*)d_out;

  // 1. weights -> f16
  cvt_f16_kernel<<<(COUT * CIN + 255) / 256, 256, 0, stream>>>(w0, W0h, COUT * CIN);
  cvt_f16_kernel<<<(COUT * COUT + 255) / 256, 256, 0, stream>>>(w1, W1h, COUT * COUT);

  // 2. points2 [256][S] -> P2t [S][256] for contiguous neighbor-row gathers
  transpose_f32_kernel<<<dim3(NS / 64, 4), 256, 0, stream>>>(points2, P2t, NS);

  // 3. masked 3-NN (LDS-staged source cloud)
  knn_kernel<<<NQ / 256, 256, 0, stream>>>(xyz1, xyz2, pid1, pid2, idxb, wgt);

  // 4. feature rows Fh[N][512] = [points1^T | interp] in f16
  transpose_f16_kernel<<<dim3(NQ / 64, 4), 256, 0, stream>>>(points1, Fh, NQ, CIN, 0);
  interp_kernel<<<NQ / 32, 256, 0, stream>>>(P2t, idxb, wgt, Fh);

  // 5. layer 1 GEMM: Y[256][N] = W0h[256][512] x Fh^T  (64-row strips per wave)
  wmma_gemm_kernel<<<dim3(NQ / 16, COUT / 64), 32, 0, stream>>>(W0h, Fh, Y, CIN, NQ);

  // 6-7. BN stats, then BN+ReLU with transpose to f16 B-operand layout
  bn_stats_kernel<<<COUT, 256, 0, stream>>>(Y, mean0, rstd0, NQ);
  bn_act_transpose_kernel<<<dim3(NQ / 64, 4), 256, 0, stream>>>(Y, mean0, rstd0, g0, bt0, X2h, NQ);

  // 8. layer 2 GEMM: Y[256][N] = W1h[256][256] x X2h^T
  wmma_gemm_kernel<<<dim3(NQ / 16, COUT / 64), 32, 0, stream>>>(W1h, X2h, Y, COUT, NQ);

  // 9-10. BN stats, then BN+ReLU straight to f32 output [1,256,N]
  bn_stats_kernel<<<COUT, 256, 0, stream>>>(Y, mean1, rstd1, NQ);
  bn_act_out_kernel<<<(COUT * NQ) / 256, 256, 0, stream>>>(Y, mean1, rstd1, g1, bt1, outF);
}
